// CrossAttentionBlock_83631603187809
// MI455X (gfx1250) — matrix operand order repaired and checked by reading, not compiled, not run on hardware
//
#include <hip/hip_runtime.h>
#include <cstdint>

// ---------------------------------------------------------------------------
// Problem constants (Swin cross-attention block)
// ---------------------------------------------------------------------------
#define BQ      16
#define CQ      256
#define CK      512
#define HW      3136          // 56*56
#define NROWS   50176         // B * HW
#define NHEAD   8
#define HD      32
#define WSZ     49            // 7*7 window
#define NWIN    1024          // B * 64 windows

#define USE_ASYNC_LDS 1       // gfx1250 GLOBAL_LOAD_ASYNC_TO_LDS_B128 staging

typedef __attribute__((ext_vector_type(16))) _Float16 v16h;
typedef __attribute__((ext_vector_type(8)))  float    v8f;

// ---------------------------------------------------------------------------
// WMMA fragment helpers (ISA 7.12.2 layouts, wave32)
//   A (16x32 f16): lane L -> row = L&15; hi = L>>4 selects K groups
//                  e[0..7]  = row[hi*8 .. hi*8+7]
//                  e[8..15] = row[16+hi*8 .. 16+hi*8+7]
//   B (32x16 f16): lane L -> col = L&15; hi = L>>4 selects K groups (same as A)
//   C/D (16x16 f32): lane L -> col = L&15; vgpr v -> row = v + 8*(L>>4)
// ---------------------------------------------------------------------------
static __device__ inline v16h load_a16(const _Float16* rowPtr, int hi) {
  union { v16h v; uint4 q[2]; } u;
  u.q[0] = *(const uint4*)(rowPtr + hi * 8);
  u.q[1] = *(const uint4*)(rowPtr + 16 + hi * 8);
  return u.v;
}
static __device__ inline v16h load_b16(const _Float16* p, int ld, int hi) {
  v16h v;
#pragma unroll
  for (int s = 0; s < 8; ++s) {
    v[s]     = p[(hi * 8 + s) * ld];
    v[s + 8] = p[(16 + hi * 8 + s) * ld];
  }
  return v;
}
static __device__ inline v8f wmma_f16(v16h a, v16h b, v8f c) {
  return __builtin_amdgcn_wmma_f32_16x16x32_f16(false, a, false, b,
                                                (short)0, c, false, false);
}

// Single-instruction XOR lane swizzle (group-of-32 mode, and_mask=0x1F).
// All masks used are <16 so 16-lane groups are preserved.
template <int MASK>
static __device__ inline float swz_xor(float x) {
  return __int_as_float(
      __builtin_amdgcn_ds_swizzle(__float_as_int(x), (MASK << 10) | 0x1f));
}

// Async global -> LDS 16-byte copy (ASYNCcnt-tracked, CDNA5).
static __device__ inline void async_copy16(void* ldsDst, const void* gSrc) {
#if USE_ASYNC_LDS
  unsigned la = (unsigned)(uintptr_t)ldsDst;
  asm volatile("global_load_async_to_lds_b128 %0, %1, off"
               :: "v"(la), "v"(gSrc) : "memory");
#else
  *(uint4*)ldsDst = *(const uint4*)gSrc;
#endif
}
static __device__ inline void async_wait() {
#if USE_ASYNC_LDS
  asm volatile("s_wait_asynccnt 0x0" ::: "memory");
#endif
}

// ---------------------------------------------------------------------------
// f32 -> f16 weight conversion
// ---------------------------------------------------------------------------
__global__ __launch_bounds__(256) void cvt_f32_f16(const float* __restrict__ src,
                                                   _Float16* __restrict__ dst,
                                                   int n) {
  int i = blockIdx.x * 256 + threadIdx.x;
  if (i < n) dst[i] = (_Float16)src[i];
}

// ---------------------------------------------------------------------------
// LayerNorm over channel dim. Input layout [B, C, H*W]; one wave per token.
// Emits f16 row-major [row][C] (GEMM operand) and optional f32 copy (residual).
// ---------------------------------------------------------------------------
template <int C>
__global__ __launch_bounds__(256) void ln_rows(const float* __restrict__ x,
                                               const float* __restrict__ w,
                                               const float* __restrict__ bvec,
                                               _Float16* __restrict__ o16,
                                               float* __restrict__ o32) {
  int row  = blockIdx.x * 8 + (threadIdx.x >> 5);
  int lane = threadIdx.x & 31;
  int b  = row / HW;
  int hw = row - b * HW;
  const float* px = x + (size_t)b * C * HW + hw;   // channel stride = HW

  constexpr int NPER = C / 32;
  float vals[NPER];
  float s = 0.f, s2 = 0.f;
#pragma unroll
  for (int i = 0; i < NPER; ++i) {
    int ch = lane + (i << 5);
    float t = px[(size_t)ch * HW];
    vals[i] = t; s += t; s2 += t * t;
  }
#pragma unroll
  for (int o = 1; o < 32; o <<= 1) {
    s  += __shfl_xor(s,  o, 32);
    s2 += __shfl_xor(s2, o, 32);
  }
  float mean = s * (1.0f / C);
  float var  = s2 * (1.0f / C) - mean * mean;
  float rs   = rsqrtf(var + 1e-5f);
#pragma unroll
  for (int i = 0; i < NPER; ++i) {
    int ch = lane + (i << 5);
    float y = (vals[i] - mean) * rs * w[ch] + bvec[ch];
    o16[(size_t)row * C + ch] = (_Float16)y;
    if (o32) o32[(size_t)row * C + ch] = y;
  }
}

// ---------------------------------------------------------------------------
// LDS-tiled WMMA GEMM: C[M=50176, N=256] = A[M,K](f16) * W[K,N](f16) + bias
// Block: 256 threads (8 waves), tile BM=128 x BN=64, BK=32.
// Tiles staged with GLOBAL_LOAD_ASYNC_TO_LDS_B128 (ASYNCcnt).
// EPI 0: Q proj  -> scale 1/sqrt(HD), write window/head layout f16
// EPI 1: K proj  -> write window/head layout f16
// EPI 2: V proj  -> write window/head layout f16
// EPI 3: Wo proj -> reshape-as-(B,H,W,C) + bias + residual(qs f32) -> d_out
// ---------------------------------------------------------------------------
template <int EPI>
__global__ __launch_bounds__(256) void gemm_wmma(const _Float16* __restrict__ A,
                                                 const _Float16* __restrict__ W,
                                                 const float* __restrict__ bias,
                                                 int Kdim,
                                                 _Float16* __restrict__ o16,
                                                 float* __restrict__ o32,
                                                 const float* __restrict__ resid) {
  __shared__ _Float16 As[128 * 32];   // 8 KB
  __shared__ _Float16 Bs[32 * 64];    // 4 KB

  int tid  = threadIdx.x;
  int lane = tid & 31;
  int wav  = tid >> 5;
  int nl   = lane & 15;
  int hi   = lane >> 4;
  int mBase = blockIdx.y * 128;
  int nBase = blockIdx.x * 64;
  int m0w   = wav * 16;

  v8f z = {0.f, 0.f, 0.f, 0.f, 0.f, 0.f, 0.f, 0.f};
  v8f acc[4] = {z, z, z, z};

  int nk = Kdim >> 5;
  for (int kt = 0; kt < nk; ++kt) {
    // stage A tile: 128x32 f16, two 16B async transfers per thread
#pragma unroll
    for (int p = 0; p < 2; ++p) {
      int j = p * 2048 + tid * 8;
      int row = j >> 5, col = j & 31;
      async_copy16(As + j, A + (size_t)(mBase + row) * Kdim + kt * 32 + col);
    }
    // stage B tile: 32x64 f16, one 16B async transfer per thread
    {
      int j = tid * 8;
      int row = j >> 6, col = j & 63;
      async_copy16(Bs + j, W + (size_t)(kt * 32 + row) * 256 + nBase + col);
    }
    async_wait();
    __syncthreads();

    v16h a = load_a16(As + (m0w + nl) * 32, hi);
#pragma unroll
    for (int nt = 0; nt < 4; ++nt) {
      v16h bf = load_b16(Bs + nt * 16 + nl, 64, hi);
      acc[nt] = wmma_f16(a, bf, acc[nt]);
    }
    __syncthreads();
  }

  // ---- fused epilogue ----
#pragma unroll
  for (int nt = 0; nt < 4; ++nt) {
#pragma unroll
    for (int v = 0; v < 8; ++v) {
      int m = mBase + m0w + v + 8 * hi;          // token row
      int n = nBase + nt * 16 + nl;              // output channel
      float val = acc[nt][v] + bias[n];
      if (EPI == 0) val *= 0.1767766952966369f;  // HD^-0.5
      int b  = m / HW;
      int m2 = m - b * HW;
      if (EPI <= 2) {
        int r = m2 / 56, c = m2 - r * 56;
        int win  = b * 64 + (r / 7) * 8 + (c / 7);
        int pos  = (r % 7) * 7 + (c % 7);
        int head = n >> 5, d = n & 31;
        o16[(((size_t)(win * 8 + head)) * WSZ + pos) * HD + d] = (_Float16)val;
      } else {
        // reference reshapes window-ordered [Bw,49,C] directly as [B,H,W,C];
        // residual row index in natural order equals m exactly.
        val += resid[(size_t)m * 256 + n];
        o32[((size_t)(b * 256 + n)) * HW + m2] = val;
      }
    }
  }
}

// ---------------------------------------------------------------------------
// Windowed attention: one block (128 thr = 4 waves) per (window, head).
// S = Q K^T (+rel-pos bias), softmax, O = P V.  All matmuls via WMMA f16.
// Bias is staged as a padded 64x64 tile so the softmax path is branch-free:
//   n >= 49            -> -1e30 (masked keys)
//   m >= 49, n < 49    -> 0     (padded query rows stay finite)
// ---------------------------------------------------------------------------
__global__ __launch_bounds__(128) void attn_wmma(const _Float16* __restrict__ Qw,
                                                 const _Float16* __restrict__ Kw,
                                                 const _Float16* __restrict__ Vw,
                                                 const float* __restrict__ rpb,
                                                 _Float16* __restrict__ AO) {
  __shared__ _Float16 Qs[64 * 32];     // rows padded 49->64 with zeros
  __shared__ _Float16 Kt[32 * 64];     // K transposed: [d][pos]
  __shared__ _Float16 Vs[64 * 32];
  __shared__ _Float16 Ps[64 * 64];     // softmax(P) staged for second GEMM
  __shared__ float    BiasP[64 * 64];  // padded rel-pos bias + mask

  int bx   = blockIdx.x;
  int win  = bx >> 3;
  int head = bx & 7;
  int tid  = threadIdx.x;
  int lane = tid & 31;
  int wav  = tid >> 5;
  int nl   = lane & 15;
  int hi   = lane >> 4;

  size_t base = ((size_t)(win * 8 + head)) * (WSZ * HD);

  for (int idx = tid; idx < 64 * 32; idx += 128) {
    int pos = idx >> 5, d = idx & 31;
    bool ok = pos < WSZ;
    _Float16 qv = ok ? Qw[base + pos * HD + d] : (_Float16)0.0f;
    _Float16 kv = ok ? Kw[base + pos * HD + d] : (_Float16)0.0f;
    _Float16 vv = ok ? Vw[base + pos * HD + d] : (_Float16)0.0f;
    Qs[idx] = qv;
    Kt[d * 64 + pos] = kv;
    Vs[idx] = vv;
  }
  // padded relative-position bias tile (static Swin index, computed inline)
  for (int idx = tid; idx < 64 * 64; idx += 128) {
    int m = idx >> 6, n = idx & 63;
    float bv;
    if (n >= WSZ)      bv = -1e30f;
    else if (m < WSZ) {
      int ph = m / 7, pw = m - ph * 7;
      int qh = n / 7, qw = n - qh * 7;
      int ridx = (ph - qh + 6) * 13 + (pw - qw + 6);
      bv = rpb[ridx * NHEAD + head];
    } else             bv = 0.0f;
    BiasP[idx] = bv;
  }
  __syncthreads();

  int m0 = wav * 16;   // this wave owns output rows m0..m0+15

  // ---- S = Q K^T : 4 N-tiles, single K-step (HD = 32) ----
  v8f z = {0.f, 0.f, 0.f, 0.f, 0.f, 0.f, 0.f, 0.f};
  v8f acc[4] = {z, z, z, z};
  {
    v16h a = load_a16(Qs + (m0 + nl) * 32, hi);
#pragma unroll
    for (int nt = 0; nt < 4; ++nt) {
      v16h bf = load_b16(Kt + nt * 16 + nl, 64, hi);
      acc[nt] = wmma_f16(a, bf, acc[nt]);
    }
  }

  // ---- bias + masked softmax (branch-free, ds_swizzle reductions) ----
#pragma unroll
  for (int v = 0; v < 8; ++v) {
    int m = m0 + v + 8 * hi;
    const float* brow = BiasP + m * 64 + nl;
    float rm = -1e30f;
#pragma unroll
    for (int nt = 0; nt < 4; ++nt) {
      float s = acc[nt][v] + brow[nt * 16];
      acc[nt][v] = s;
      rm = fmaxf(rm, s);
    }
    rm = fmaxf(rm, swz_xor<1>(rm));
    rm = fmaxf(rm, swz_xor<2>(rm));
    rm = fmaxf(rm, swz_xor<4>(rm));
    rm = fmaxf(rm, swz_xor<8>(rm));
    float sum = 0.f;
#pragma unroll
    for (int nt = 0; nt < 4; ++nt) {
      float p = __expf(acc[nt][v] - rm);
      acc[nt][v] = p;
      sum += p;
    }
    sum += swz_xor<1>(sum);
    sum += swz_xor<2>(sum);
    sum += swz_xor<4>(sum);
    sum += swz_xor<8>(sum);
    float inv = 1.0f / sum;
#pragma unroll
    for (int nt = 0; nt < 4; ++nt) {
      int n = nt * 16 + nl;
      Ps[m * 64 + n] = (_Float16)(acc[nt][v] * inv);
    }
  }
  // each wave reads back only its own P rows: same-wave LDS ordering suffices

  // ---- O = P V : 2 N-tiles (d 0..31), 2 K-steps over 64 padded keys ----
  v8f o0 = z, o1 = z;
#pragma unroll
  for (int ks = 0; ks < 2; ++ks) {
    v16h ap = load_a16(Ps + (m0 + nl) * 64 + ks * 32, hi);
    v16h b0 = load_b16(Vs + (ks * 32) * 32 + nl, 32, hi);
    v16h b1 = load_b16(Vs + (ks * 32) * 32 + 16 + nl, 32, hi);
    o0 = wmma_f16(ap, b0, o0);
    o1 = wmma_f16(ap, b1, o1);
  }
#pragma unroll
  for (int v = 0; v < 8; ++v) {
    int m = m0 + v + 8 * hi;
    if (m < WSZ) {
      size_t row = (size_t)(win * WSZ + m) * 256 + head * HD;
      AO[row + nl]      = (_Float16)o0[v];
      AO[row + 16 + nl] = (_Float16)o1[v];
    }
  }
}

// ---------------------------------------------------------------------------
// Launch
// ---------------------------------------------------------------------------
extern "C" void kernel_launch(void* const* d_in, const int* in_sizes, int n_in,
                              void* d_out, int out_size, void* d_ws, size_t ws_size,
                              hipStream_t stream) {
  (void)in_sizes; (void)n_in; (void)out_size; (void)ws_size;
  const float* q   = (const float*)d_in[0];
  const float* kv  = (const float*)d_in[1];
  const float* nqw = (const float*)d_in[2];
  const float* nqb = (const float*)d_in[3];
  const float* nkw = (const float*)d_in[4];
  const float* nkb = (const float*)d_in[5];
  const float* Wq  = (const float*)d_in[6];
  const float* bq  = (const float*)d_in[7];
  const float* Wk  = (const float*)d_in[8];
  const float* bk  = (const float*)d_in[9];
  const float* Wv  = (const float*)d_in[10];
  const float* bv  = (const float*)d_in[11];
  const float* Wo  = (const float*)d_in[12];
  const float* bo  = (const float*)d_in[13];
  const float* rpb = (const float*)d_in[14];
  float* out = (float*)d_out;

  char* ws = (char*)d_ws;
  size_t off = 0;
  auto take = [&](size_t bytes) -> char* {
    char* p = ws + off;
    off = (off + bytes + 255) & ~(size_t)255;
    return p;
  };
  _Float16* qf16  = (_Float16*)take((size_t)NROWS * CQ * 2);
  float*    qs32  = (float*)   take((size_t)NROWS * CQ * 4);
  _Float16* kvf16 = (_Float16*)take((size_t)NROWS * CK * 2);
  _Float16* Wq16  = (_Float16*)take((size_t)CQ * CQ * 2);
  _Float16* Wk16  = (_Float16*)take((size_t)CK * CQ * 2);
  _Float16* Wv16  = (_Float16*)take((size_t)CK * CQ * 2);
  _Float16* Wo16  = (_Float16*)take((size_t)CQ * CQ * 2);
  _Float16* Qwb   = (_Float16*)take((size_t)NROWS * CQ * 2);
  _Float16* Kwb   = (_Float16*)take((size_t)NROWS * CQ * 2);
  _Float16* Vwb   = (_Float16*)take((size_t)NROWS * CQ * 2);
  _Float16* AO    = (_Float16*)take((size_t)NROWS * CQ * 2);

  // weight conversion f32 -> f16
  cvt_f32_f16<<<(CQ * CQ + 255) / 256, 256, 0, stream>>>(Wq, Wq16, CQ * CQ);
  cvt_f32_f16<<<(CK * CQ + 255) / 256, 256, 0, stream>>>(Wk, Wk16, CK * CQ);
  cvt_f32_f16<<<(CK * CQ + 255) / 256, 256, 0, stream>>>(Wv, Wv16, CK * CQ);
  cvt_f32_f16<<<(CQ * CQ + 255) / 256, 256, 0, stream>>>(Wo, Wo16, CQ * CQ);

  // layernorms (one wave per token, 8 waves per block)
  ln_rows<CQ><<<NROWS / 8, 256, 0, stream>>>(q,  nqw, nqb, qf16,  qs32);
  ln_rows<CK><<<NROWS / 8, 256, 0, stream>>>(kv, nkw, nkb, kvf16, nullptr);

  // projections (fused bias / scale / window-head relayout)
  dim3 ggrid(CQ / 64, NROWS / 128);   // (4, 392)
  gemm_wmma<0><<<ggrid, 256, 0, stream>>>(qf16,  Wq16, bq, CQ, Qwb, nullptr, nullptr);
  gemm_wmma<1><<<ggrid, 256, 0, stream>>>(kvf16, Wk16, bk, CK, Kwb, nullptr, nullptr);
  gemm_wmma<2><<<ggrid, 256, 0, stream>>>(kvf16, Wv16, bv, CK, Vwb, nullptr, nullptr);

  // windowed attention: 1024 windows x 8 heads
  attn_wmma<<<NWIN * NHEAD, 128, 0, stream>>>(Qwb, Kwb, Vwb, rpb, AO);

  // output projection + reshape + residual -> d_out
  gemm_wmma<3><<<ggrid, 256, 0, stream>>>(AO, Wo16, bo, CQ, nullptr, out, qs32);
}
